// HyperGATLayer_76124000354328
// MI455X (gfx1250) — compile-verified, matrix operand order checked by reading, not run on hardware
//
// HyperGAT layer for MI455X (gfx1250, wave32, WMMA bf16 16x16x32, fp32 accum).
//
// Pipeline (all on `stream`, ~40 MB workspace):
//  1. Wn,We -> bf16, pre-transposed [c][k]           (tiny)
//  2. NPt aux rows: row512=1.0, rows 513..527=0      (tiny)
//  3. node_proj^T (bf16 [528][N]) = X @ Wn + bn       WMMA
//  4. edge_acc [E][528] = H_inc^T @ node_proj_ext     WMMA (col512 = deg_e), A via LDS transpose
//  5. edge_emb [E][512] = (edge_acc/deg) @ We + be    WMMA
//  6. scores[e] = edge_emb @ Wa + ba                  wave-per-row dot
//  7. global max of scores                            1 WG reduce
//  8. Bt [528][E] bf16 = (exp(s-M) * edge_emb)^T, col512 = exp(s-M)
//  9. out = LN( (H_inc @ Bt^T)[:, :512] / den + X )   WMMA + fused LDS LayerNorm
#include <hip/hip_runtime.h>

typedef __bf16 bf16_t;
typedef __attribute__((ext_vector_type(16))) __bf16 v16bf;
typedef __attribute__((ext_vector_type(8)))  float  v8f;

#define N_TOT 16384
#define E_TOT 4096
#define HID   512
#define COLS  528   // 512 + aux col (512 = ones / softmax weight) + 15 zero cols

union BFrag { v16bf v; uint4 q[2]; };
union Pack8 { uint4 q; bf16_t h[8]; };

__device__ __forceinline__ v8f wmma_bf16(v16bf a, v16bf b, v8f c) {
  // D = A(16x32) * B(32x16) + C, fp32 accumulate
  return __builtin_amdgcn_wmma_f32_16x16x32_bf16(false, a, false, b, (short)0, c, false, false);
}

__device__ __forceinline__ v8f vzero8() {
  v8f z;
#pragma unroll
  for (int i = 0; i < 8; ++i) z[i] = 0.f;
  return z;
}

// A-fragment from fp32 row-major row: p points at K=kk of this lane's row.
// v16bf elements 0..7 = K kk..kk+7, elements 8..15 = K kk+16..kk+23.
__device__ __forceinline__ v16bf afrag_f32(const float* p, float scale) {
  const float4 a0 = *(const float4*)(p);
  const float4 a1 = *(const float4*)(p + 4);
  const float4 a2 = *(const float4*)(p + 16);
  const float4 a3 = *(const float4*)(p + 20);
  v16bf r;
  r[0] = (bf16_t)(a0.x * scale); r[1] = (bf16_t)(a0.y * scale);
  r[2] = (bf16_t)(a0.z * scale); r[3] = (bf16_t)(a0.w * scale);
  r[4] = (bf16_t)(a1.x * scale); r[5] = (bf16_t)(a1.y * scale);
  r[6] = (bf16_t)(a1.z * scale); r[7] = (bf16_t)(a1.w * scale);
  r[8] = (bf16_t)(a2.x * scale); r[9] = (bf16_t)(a2.y * scale);
  r[10] = (bf16_t)(a2.z * scale); r[11] = (bf16_t)(a2.w * scale);
  r[12] = (bf16_t)(a3.x * scale); r[13] = (bf16_t)(a3.y * scale);
  r[14] = (bf16_t)(a3.z * scale); r[15] = (bf16_t)(a3.w * scale);
  return r;
}

// B-fragment from bf16 B^T [C][K] (K contiguous): two global_load_b128 per lane.
__device__ __forceinline__ v16bf bfrag_ld(const bf16_t* bt, long ldk, int cbase,
                                          int k0, int lrow, int lhi) {
  const bf16_t* p = bt + (long)(cbase + lrow) * ldk + k0 + lhi * 8;
  BFrag f;
  f.q[0] = *(const uint4*)(p);
  f.q[1] = *(const uint4*)(p + 16);
  return f.v;
}

// ---------------- prep kernels ----------------
__global__ __launch_bounds__(256) void k_w_to_bf16T(const float* __restrict__ W,
                                                    bf16_t* __restrict__ Wt) {
  int idx = blockIdx.x * 256 + threadIdx.x;  // HID*HID threads
  int k = idx & (HID - 1);
  int c = idx >> 9;
  Wt[(long)c * HID + k] = (bf16_t)W[(long)k * HID + c];
}

__global__ __launch_bounds__(256) void k_npt_aux(bf16_t* __restrict__ NPt) {
  int idx = blockIdx.x * 256 + threadIdx.x;  // 16*N_TOT threads
  int r = HID + (idx >> 14);
  int n = idx & (N_TOT - 1);
  NPt[(long)r * N_TOT + n] = (bf16_t)((r == HID) ? 1.0f : 0.0f);
}

// ---------------- pass A: node_proj^T = (X @ Wn + bn)^T, bf16 ----------------
__global__ __launch_bounds__(256) void k_node_proj(const float* __restrict__ X,
                                                   const bf16_t* __restrict__ Wnt,
                                                   const float* __restrict__ bn,
                                                   bf16_t* __restrict__ NPt) {
  const int lane = threadIdx.x & 31, wave = threadIdx.x >> 5;
  const int lrow = lane & 15, lhi = lane >> 4;
  const int rt = wave >> 2, cg = wave & 3;   // 2 row-tiles x 4 col-groups
  const int nb = blockIdx.x * 32;
  const int cb = cg * 128;
  v8f acc[8];
#pragma unroll
  for (int i = 0; i < 8; ++i) acc[i] = vzero8();
  const int arow = nb + rt * 16 + lrow;
  for (int k0 = 0; k0 < HID; k0 += 32) {
    v16bf af = afrag_f32(X + (long)arow * HID + k0 + lhi * 8, 1.0f);
#pragma unroll
    for (int ct = 0; ct < 8; ++ct) {
      v16bf bf_ = bfrag_ld(Wnt, HID, cb + ct * 16, k0, lrow, lhi);
      acc[ct] = wmma_bf16(af, bf_, acc[ct]);
    }
  }
  // C-tile: lane holds (n = nb+rt*16+lhi*8+r, c = cb+ct*16+lrow); n consecutive
  // -> one 16B bf16 store per tile directly into the transposed layout.
#pragma unroll
  for (int ct = 0; ct < 8; ++ct) {
    const int c = cb + ct * 16 + lrow;
    const float bias = bn[c];
    Pack8 pk;
#pragma unroll
    for (int r = 0; r < 8; ++r) pk.h[r] = (bf16_t)(acc[ct][r] + bias);
    *(uint4*)(NPt + (long)c * N_TOT + nb + rt * 16 + lhi * 8) = pk.q;
  }
}

// ---------------- pass B: edge_acc = H_inc^T @ node_proj_ext ----------------
#define LDSA 40  // bf16 elements per LDS row (32 + pad), 16B-aligned fragments
__global__ __launch_bounds__(256) void k_spmm_edge(const float* __restrict__ Hinc,
                                                   const bf16_t* __restrict__ NPt,
                                                   float* __restrict__ edge_acc) {
  __shared__ bf16_t ldsA[32 * LDSA];  // 32 e-rows x 32 n (transposed tile)
  const int lane = threadIdx.x & 31, wave = threadIdx.x >> 5;
  const int lrow = lane & 15, lhi = lane >> 4;
  const int rt = wave >> 2, cg = wave & 3;
  const int e0 = blockIdx.x * 32;
  const int cb = cg * 128;
  v8f acc[8];
  v8f accD = vzero8();
#pragma unroll
  for (int i = 0; i < 8; ++i) acc[i] = vzero8();
  const int tn = threadIdx.x >> 3;       // k (node) row within tile, 0..31
  const int tj = (threadIdx.x & 7) * 4;  // e offset within tile
  for (int k0 = 0; k0 < N_TOT; k0 += 32) {
    // cooperative transpose: H_inc[k0+tn][e0+tj..+3] -> ldsA[e][n], fp32->bf16
    float4 f = *(const float4*)(Hinc + (long)(k0 + tn) * E_TOT + e0 + tj);
    ldsA[(tj + 0) * LDSA + tn] = (bf16_t)f.x;
    ldsA[(tj + 1) * LDSA + tn] = (bf16_t)f.y;
    ldsA[(tj + 2) * LDSA + tn] = (bf16_t)f.z;
    ldsA[(tj + 3) * LDSA + tn] = (bf16_t)f.w;
    __syncthreads();
    BFrag af;
    af.q[0] = *(const uint4*)&ldsA[(rt * 16 + lrow) * LDSA + lhi * 8];
    af.q[1] = *(const uint4*)&ldsA[(rt * 16 + lrow) * LDSA + 16 + lhi * 8];
#pragma unroll
    for (int ct = 0; ct < 8; ++ct) {
      v16bf bf_ = bfrag_ld(NPt, N_TOT, cb + ct * 16, k0, lrow, lhi);
      acc[ct] = wmma_bf16(af.v, bf_, acc[ct]);
    }
    if (cg == 0) {  // wave-uniform: aux tile -> deg_e in col 512
      v16bf bf_ = bfrag_ld(NPt, N_TOT, HID, k0, lrow, lhi);
      accD = wmma_bf16(af.v, bf_, accD);
    }
    __syncthreads();
  }
#pragma unroll
  for (int ct = 0; ct < 8; ++ct) {
    const int c = cb + ct * 16 + lrow;
#pragma unroll
    for (int r = 0; r < 8; ++r)
      edge_acc[(long)(e0 + rt * 16 + lhi * 8 + r) * COLS + c] = acc[ct][r];
  }
  if (cg == 0) {
    const int c = HID + lrow;
#pragma unroll
    for (int r = 0; r < 8; ++r)
      edge_acc[(long)(e0 + rt * 16 + lhi * 8 + r) * COLS + c] = accD[r];
  }
}

// ---------------- pass C: edge_emb = (edge_acc/deg) @ We + be ----------------
__global__ __launch_bounds__(256) void k_edge_emb(const float* __restrict__ edge_acc,
                                                  const bf16_t* __restrict__ Wet,
                                                  const float* __restrict__ be,
                                                  float* __restrict__ edge_emb) {
  const int lane = threadIdx.x & 31, wave = threadIdx.x >> 5;
  const int lrow = lane & 15, lhi = lane >> 4;
  const int eb = blockIdx.x * 16;
  const int cb = wave * 64;  // 8 waves x 64 cols
  const int e = eb + lrow;
  const float deg = edge_acc[(long)e * COLS + HID];
  const float rdeg = 1.0f / (deg + 1e-8f);
  v8f acc[4];
#pragma unroll
  for (int i = 0; i < 4; ++i) acc[i] = vzero8();
  for (int k0 = 0; k0 < HID; k0 += 32) {
    v16bf af = afrag_f32(edge_acc + (long)e * COLS + k0 + lhi * 8, rdeg);
#pragma unroll
    for (int ct = 0; ct < 4; ++ct) {
      v16bf bf_ = bfrag_ld(Wet, HID, cb + ct * 16, k0, lrow, lhi);
      acc[ct] = wmma_bf16(af, bf_, acc[ct]);
    }
  }
#pragma unroll
  for (int ct = 0; ct < 4; ++ct) {
    const int c = cb + ct * 16 + lrow;
    const float bias = be[c];
#pragma unroll
    for (int r = 0; r < 8; ++r)
      edge_emb[(long)(eb + lhi * 8 + r) * HID + c] = acc[ct][r] + bias;
  }
}

// ---------------- scores, max, Bt build ----------------
__global__ __launch_bounds__(256) void k_scores(const float* __restrict__ edge_emb,
                                                const float* __restrict__ Wa,
                                                const float* __restrict__ ba,
                                                float* __restrict__ scores) {
  const int lane = threadIdx.x & 31, wave = threadIdx.x >> 5;
  const int e = blockIdx.x * 8 + wave;
  const float* row = edge_emb + (long)e * HID + lane * 16;
  const float* wv = Wa + lane * 16;
  float s = 0.f;
#pragma unroll
  for (int i = 0; i < 16; ++i) s += row[i] * wv[i];
#pragma unroll
  for (int off = 16; off > 0; off >>= 1) s += __shfl_xor(s, off, 32);
  if (lane == 0) scores[e] = s + ba[0];
}

__global__ __launch_bounds__(256) void k_smax(const float* __restrict__ scores,
                                              float* __restrict__ smax) {
  __shared__ float red[256];
  float m = -3.4e38f;
  for (int i = threadIdx.x; i < E_TOT; i += 256) m = fmaxf(m, scores[i]);
  red[threadIdx.x] = m;
  __syncthreads();
  for (int s = 128; s > 0; s >>= 1) {
    if (threadIdx.x < s) red[threadIdx.x] = fmaxf(red[threadIdx.x], red[threadIdx.x + s]);
    __syncthreads();
  }
  if (threadIdx.x == 0) smax[0] = red[0];
}

__global__ __launch_bounds__(256) void k_build_bt(const float* __restrict__ edge_emb,
                                                  const float* __restrict__ scores,
                                                  const float* __restrict__ smax,
                                                  bf16_t* __restrict__ Bt) {
  const int e = blockIdx.x * 256 + threadIdx.x;
  const int c = blockIdx.y;
  const float w = __expf(scores[e] - smax[0]);  // softmax weight, shift-invariant
  float v;
  if (c < HID)       v = w * edge_emb[(long)e * HID + c];
  else if (c == HID) v = w;   // denominator column
  else               v = 0.f;
  Bt[(long)c * E_TOT + e] = (bf16_t)v;
}

// ------- pass D: out = LayerNorm( (H_inc @ Bt^T)/den + X ) * gamma + beta -------
__global__ __launch_bounds__(256) void k_attn_out_ln(const float* __restrict__ Hinc,
                                                     const bf16_t* __restrict__ Bt,
                                                     const float* __restrict__ X,
                                                     const float* __restrict__ g,
                                                     const float* __restrict__ b,
                                                     float* __restrict__ out) {
  extern __shared__ float smem[];  // [32*512] numerator tile + [32] denominators
  float* sy = smem;
  float* sden = smem + 32 * HID;
  const int lane = threadIdx.x & 31, wave = threadIdx.x >> 5;
  const int lrow = lane & 15, lhi = lane >> 4;
  const int rt = wave >> 2, cg = wave & 3;
  const int nb = blockIdx.x * 32;
  const int cb = cg * 128;
  v8f acc[8];
  v8f accD = vzero8();
#pragma unroll
  for (int i = 0; i < 8; ++i) acc[i] = vzero8();
  const int arow = nb + rt * 16 + lrow;
  for (int k0 = 0; k0 < E_TOT; k0 += 32) {
    v16bf af = afrag_f32(Hinc + (long)arow * E_TOT + k0 + lhi * 8, 1.0f);
#pragma unroll
    for (int ct = 0; ct < 8; ++ct) {
      v16bf bf_ = bfrag_ld(Bt, E_TOT, cb + ct * 16, k0, lrow, lhi);
      acc[ct] = wmma_bf16(af, bf_, acc[ct]);
    }
    if (cg == 0) {  // denominator tile (col 512)
      v16bf bf_ = bfrag_ld(Bt, E_TOT, HID, k0, lrow, lhi);
      accD = wmma_bf16(af, bf_, accD);
    }
  }
#pragma unroll
  for (int ct = 0; ct < 8; ++ct) {
    const int c = cb + ct * 16 + lrow;
#pragma unroll
    for (int r = 0; r < 8; ++r) sy[(rt * 16 + lhi * 8 + r) * HID + c] = acc[ct][r];
  }
  if (cg == 0 && lrow == 0) {
#pragma unroll
    for (int r = 0; r < 8; ++r) sden[rt * 16 + lhi * 8 + r] = accD[r];
  }
  __syncthreads();
  // Fused residual + LayerNorm: each wave owns 4 rows, lane covers 16 contiguous cols.
#pragma unroll
  for (int rr = 0; rr < 4; ++rr) {
    const int r = wave * 4 + rr;
    const int n = nb + r;
    const float den = sden[r];
    const float scale = (den > 0.f) ? (1.0f / den) : 0.0f;  // no incident edges -> 0
    float y[16];
    float s = 0.f, s2 = 0.f;
#pragma unroll
    for (int i = 0; i < 16; ++i) {
      const int c = lane * 16 + i;
      const float v = sy[r * HID + c] * scale + X[(long)n * HID + c];
      y[i] = v; s += v; s2 += v * v;
    }
#pragma unroll
    for (int off = 16; off > 0; off >>= 1) {
      s += __shfl_xor(s, off, 32);
      s2 += __shfl_xor(s2, off, 32);
    }
    const float mu = s * (1.0f / HID);
    const float var = s2 * (1.0f / HID) - mu * mu;
    const float rstd = rsqrtf(var + 1e-5f);
#pragma unroll
    for (int i = 0; i < 16; ++i) {
      const int c = lane * 16 + i;
      out[(long)n * HID + c] = g[c] * (y[i] - mu) * rstd + b[c];
    }
  }
}

extern "C" void kernel_launch(void* const* d_in, const int* in_sizes, int n_in,
                              void* d_out, int out_size, void* d_ws, size_t ws_size,
                              hipStream_t stream) {
  (void)in_sizes; (void)n_in; (void)out_size; (void)ws_size;
  const float* node_emb = (const float*)d_in[0];
  const float* Hinc     = (const float*)d_in[1];
  const float* Wn       = (const float*)d_in[2];
  const float* bn       = (const float*)d_in[3];
  const float* We       = (const float*)d_in[4];
  const float* be       = (const float*)d_in[5];
  const float* Wa       = (const float*)d_in[6];
  const float* ba       = (const float*)d_in[7];
  const float* gamma_   = (const float*)d_in[8];
  const float* beta_    = (const float*)d_in[9];
  float* out = (float*)d_out;

  char* ws = (char*)d_ws;
  size_t off = 0;
  auto carve = [&](size_t bytes) -> void* {
    void* p = ws + off;
    off += (bytes + 255) & ~(size_t)255;
    return p;
  };
  bf16_t* NPt      = (bf16_t*)carve((size_t)COLS * N_TOT * sizeof(bf16_t));  // 17.3 MB
  float*  edge_acc = (float*) carve((size_t)E_TOT * COLS * sizeof(float));   //  8.7 MB
  float*  edge_emb = (float*) carve((size_t)E_TOT * HID * sizeof(float));    //  8.4 MB
  float*  scores   = (float*) carve((size_t)E_TOT * sizeof(float));
  float*  smax     = (float*) carve(256);
  bf16_t* Bt       = (bf16_t*)carve((size_t)COLS * E_TOT * sizeof(bf16_t));  //  4.3 MB
  bf16_t* Wnt      = (bf16_t*)carve((size_t)HID * HID * sizeof(bf16_t));
  bf16_t* Wet      = (bf16_t*)carve((size_t)HID * HID * sizeof(bf16_t));

  k_w_to_bf16T<<<HID * HID / 256, 256, 0, stream>>>(Wn, Wnt);
  k_w_to_bf16T<<<HID * HID / 256, 256, 0, stream>>>(We, Wet);
  k_npt_aux<<<16 * N_TOT / 256, 256, 0, stream>>>(NPt);
  k_node_proj<<<N_TOT / 32, 256, 0, stream>>>(node_emb, Wnt, bn, NPt);
  k_spmm_edge<<<E_TOT / 32, 256, 0, stream>>>(Hinc, NPt, edge_acc);
  k_edge_emb<<<E_TOT / 16, 256, 0, stream>>>(edge_acc, Wet, be, edge_emb);
  k_scores<<<E_TOT / 8, 256, 0, stream>>>(edge_emb, Wa, ba, scores);
  k_smax<<<1, 256, 0, stream>>>(scores, smax);
  k_build_bt<<<dim3(E_TOT / 256, COLS), 256, 0, stream>>>(edge_emb, scores, smax, Bt);
  k_attn_out_ln<<<N_TOT / 32, 256, (32 * HID + 32) * sizeof(float), stream>>>(
      Hinc, Bt, node_emb, gamma_, beta_, out);
}
